// CompositeGNNnodeBased_88347477279262
// MI455X (gfx1250) — matrix operand order repaired and checked by reading, not compiled, not run on hardware
//
#include <hip/hip_runtime.h>
#include <hip/hip_bf16.h>

// ---------------------------------------------------------------------------
// Types for CDNA5 WMMA (wave32): D(f32 16x16) = A(bf16 16x32) * B(bf16 32x16) + C
// ---------------------------------------------------------------------------
typedef __attribute__((ext_vector_type(16))) __bf16 v16bf;
typedef __attribute__((ext_vector_type(8)))  float  v8f;
typedef __attribute__((ext_vector_type(4)))  float  v4f;

union Frag {
    v16bf bf;
    v4f   f4[2];   // two 16-byte halves (8 bf16 each)
};

__device__ __forceinline__ unsigned short f2bf(float f) {
    unsigned u = __float_as_uint(f);
    u += 0x7FFFu + ((u >> 16) & 1u);          // round-to-nearest-even
    return (unsigned short)(u >> 16);
}

#define WMMA_BF16(A, B, C) \
    __builtin_amdgcn_wmma_f32_16x16x32_bf16(false, (A), false, (B), (short)0, (C), false, false)

// ---------------------------------------------------------------------------
// Elementwise / setup kernels
// ---------------------------------------------------------------------------
__global__ void k_zero_f32(float* p, int n, const int* flag) {
    if (flag && *flag == 0) return;
    int i = blockIdx.x * 256 + threadIdx.x;
    if (i < n) p[i] = 0.0f;
}

__global__ void k_zero_i32(int* p, int n) {
    int i = blockIdx.x * 256 + threadIdx.x;
    if (i < n) p[i] = 0;
}

__global__ void k_init_state(const float* s0, float* cur, float* old_, int n) {
    int i = blockIdx.x * 256 + threadIdx.x;
    if (i < n) { cur[i] = s0[i]; old_[i] = 1.0f; }
}

// Pack row-major f32 weight [K, Nn] into bf16 wave32 B-fragment order:
// P[((ntile*numKc + kc)*32 + lane)*16 + i] = W[kc*32 + (lane<16?0:8) + i + (i<8?0:8)][ntile*16 + lane%16]
__global__ void k_pack_b(const float* W, unsigned short* P, int K, int Nn, int total) {
    int idx = blockIdx.x * 256 + threadIdx.x;
    if (idx >= total) return;
    int i     = idx & 15;
    int lane  = (idx >> 4) & 31;
    int rest  = idx >> 9;
    int numKc = K >> 5;
    int kc    = rest % numKc;
    int nt    = rest / numKc;
    int k = kc * 32 + ((lane < 16) ? 0 : 8) + i + ((i < 8) ? 0 : 8);
    int n = nt * 16 + (lane & 15);
    P[idx] = f2bf(W[(size_t)k * Nn + n]);
}

// agg_nodes[t] = segment_sum(nodes[senders] masked by sender type) into aggc cols [t*128, t*128+128)
__global__ void k_scatter_comp_nodes(const float* nodes, const int* snd, const int* rcv,
                                     const int* ntype, float* aggc, int total) {
    int idx = blockIdx.x * 256 + threadIdx.x;
    if (idx >= total) return;
    int e = idx >> 7, c = idx & 127;
    int s = snd[e], r = rcv[e];
    int t = ntype[s];
    atomicAdd(&aggc[(size_t)r * 288 + t * 128 + c], nodes[(size_t)s * 128 + c]);
}

// agg_arcs = segment_sum(arcs_label) into aggc cols [256, 288)
__global__ void k_scatter_comp_arcs(const float* arcs, const int* rcv, float* aggc, int total) {
    int idx = blockIdx.x * 256 + threadIdx.x;
    if (idx >= total) return;
    int e = idx >> 5, c = idx & 31;
    atomicAdd(&aggc[(size_t)rcv[e] * 288 + 256 + c], arcs[(size_t)e * 32 + c]);
}

// Static bf16 columns of inp[N,672]: cols 0..127 = nodes, cols 384..671 = aggc
__global__ void k_build_static(const float* nodes, const float* aggc, unsigned short* inpB, int total) {
    int idx = blockIdx.x * 256 + threadIdx.x;
    if (idx >= total) return;                  // total = N*416
    int n = idx / 416, j = idx % 416;
    if (j < 128)
        inpB[(size_t)n * 672 + j]       = f2bf(nodes[(size_t)n * 128 + j]);
    else
        inpB[(size_t)n * 672 + 256 + j] = f2bf(aggc[(size_t)n * 288 + (j - 128)]);
}

// Convergence test: any node with |s - s_old| > THRESH * |s_old|  (compared squared)
__global__ void k_flag(const float* cur, const float* old_, int* flag, int N) {
    int node = blockIdx.x * 256 + threadIdx.x;
    if (node >= N) return;
    const float* c = cur  + (size_t)node * 128;
    const float* o = old_ + (size_t)node * 128;
    float d2 = 0.0f, n2 = 0.0f;
    for (int j = 0; j < 128; j += 4) {
        v4f cv = *(const v4f*)(c + j);
        v4f ov = *(const v4f*)(o + j);
        for (int k = 0; k < 4; ++k) {
            float df = cv[k] - ov[k];
            d2 += df * df;
            n2 += ov[k] * ov[k];
        }
    }
    if (d2 > 1e-4f * n2) atomicOr(flag, 1);
}

// agg_state = segment_sum(state[senders])
__global__ void k_scatter_state(const float* cur, const int* snd, const int* rcv,
                                float* aggs, int total, const int* flag) {
    if (*flag == 0) return;
    int idx = blockIdx.x * 256 + threadIdx.x;
    if (idx >= total) return;
    int e = idx >> 7, c = idx & 127;
    atomicAdd(&aggs[(size_t)rcv[e] * 128 + c], cur[(size_t)snd[e] * 128 + c]);
}

// Dynamic bf16 columns of inp: cols 128..255 = state, cols 256..383 = agg_state
__global__ void k_build_dyn(const float* cur, const float* aggs, unsigned short* inpB,
                            int total, const int* flag) {
    if (*flag == 0) return;
    int idx = blockIdx.x * 256 + threadIdx.x;
    if (idx >= total) return;                  // total = N*128
    int n = idx >> 7, c = idx & 127;
    size_t base = (size_t)n * 672;
    inpB[base + 128 + c] = f2bf(cur[idx]);
    inpB[base + 256 + c] = f2bf(aggs[idx]);
}

__global__ void k_commit(float* cur, float* old_, const float* snew, int total, const int* flag) {
    if (*flag == 0) return;
    int i = blockIdx.x * 256 + threadIdx.x;
    if (i < total) { old_[i] = cur[i]; cur[i] = snew[i]; }
}

__global__ void k_to_bf16(const float* src, unsigned short* dst, int total) {
    int i = blockIdx.x * 256 + threadIdx.x;
    if (i < total) dst[i] = f2bf(src[i]);
}

// ---------------------------------------------------------------------------
// GEMM kernels (one 16x16 output tile per wave, 4 waves per block)
// A fragment (bf16 16x32): lane holds row m = lane%16; elems 0..7 at K = k0+(lane>=16?8:0)+0..7,
//                          elems 8..15 at +16 — i.e. two b128 loads from row-major bf16.
// ---------------------------------------------------------------------------

// h_t = relu(inp @ W1_t + b1_t) for t=0,1  (K=672, N=512)
__global__ void k_gemm1(const unsigned short* inpB,
                        const unsigned short* pW10, const unsigned short* pW11,
                        const float* b10, const float* b11,
                        unsigned short* h0B, unsigned short* h1B, const int* flag) {
    if (*flag == 0) return;
    const int KC = 21;                          // 672/32
    int lane = threadIdx.x & 31, w = threadIdx.x >> 5;
    int m0 = blockIdx.x * 16;
    int nt = blockIdx.y * 4 + w;                // 0..31
    int off = (lane < 16) ? 0 : 8;
    const unsigned short* arow = inpB + (size_t)(m0 + (lane & 15)) * 672 + off;
    const unsigned short* bp0  = pW10 + (size_t)nt * KC * 512 + lane * 16;
    const unsigned short* bp1  = pW11 + (size_t)nt * KC * 512 + lane * 16;
    v8f acc0 = {}, acc1 = {};
    for (int kc = 0; kc < KC; ++kc) {
        Frag a, b0, b1;
        a.f4[0]  = *(const v4f*)(arow + kc * 32);
        a.f4[1]  = *(const v4f*)(arow + kc * 32 + 16);
        b0.f4[0] = *(const v4f*)(bp0 + (size_t)kc * 512);
        b0.f4[1] = *(const v4f*)(bp0 + (size_t)kc * 512 + 8);
        b1.f4[0] = *(const v4f*)(bp1 + (size_t)kc * 512);
        b1.f4[1] = *(const v4f*)(bp1 + (size_t)kc * 512 + 8);
        acc0 = WMMA_BF16(a.bf, b0.bf, acc0);
        acc1 = WMMA_BF16(a.bf, b1.bf, acc1);
    }
    int n = nt * 16 + (lane & 15);
    float bias0 = b10[n], bias1 = b11[n];
    int mbase = m0 + off;
    for (int r = 0; r < 8; ++r) {
        int m = mbase + r;
        float v0 = acc0[r] + bias0; v0 = v0 > 0.0f ? v0 : 0.0f;
        float v1 = acc1[r] + bias1; v1 = v1 > 0.0f ? v1 : 0.0f;
        h0B[(size_t)m * 512 + n] = f2bf(v0);
        h1B[(size_t)m * 512 + n] = f2bf(v1);
    }
}

// s_new = tanh(h_t @ W2_t + b2_t), t selected per node by node_type  (K=512, N=128)
__global__ void k_gemm2(const unsigned short* h0B, const unsigned short* h1B,
                        const unsigned short* pW20, const unsigned short* pW21,
                        const float* b20, const float* b21, const int* ntype,
                        float* snew, const int* flag) {
    if (*flag == 0) return;
    const int KC = 16;                          // 512/32
    int lane = threadIdx.x & 31, w = threadIdx.x >> 5;
    int m0 = blockIdx.x * 16;
    int nt = blockIdx.y * 4 + w;                // 0..7
    int off = (lane < 16) ? 0 : 8;
    const unsigned short* a0row = h0B + (size_t)(m0 + (lane & 15)) * 512 + off;
    const unsigned short* a1row = h1B + (size_t)(m0 + (lane & 15)) * 512 + off;
    const unsigned short* bp0   = pW20 + (size_t)nt * KC * 512 + lane * 16;
    const unsigned short* bp1   = pW21 + (size_t)nt * KC * 512 + lane * 16;
    v8f acc0 = {}, acc1 = {};
    for (int kc = 0; kc < KC; ++kc) {
        Frag a0, a1, b0, b1;
        a0.f4[0] = *(const v4f*)(a0row + kc * 32);
        a0.f4[1] = *(const v4f*)(a0row + kc * 32 + 16);
        a1.f4[0] = *(const v4f*)(a1row + kc * 32);
        a1.f4[1] = *(const v4f*)(a1row + kc * 32 + 16);
        b0.f4[0] = *(const v4f*)(bp0 + (size_t)kc * 512);
        b0.f4[1] = *(const v4f*)(bp0 + (size_t)kc * 512 + 8);
        b1.f4[0] = *(const v4f*)(bp1 + (size_t)kc * 512);
        b1.f4[1] = *(const v4f*)(bp1 + (size_t)kc * 512 + 8);
        acc0 = WMMA_BF16(a0.bf, b0.bf, acc0);
        acc1 = WMMA_BF16(a1.bf, b1.bf, acc1);
    }
    int n = nt * 16 + (lane & 15);
    float bias0 = b20[n], bias1 = b21[n];
    int mbase = m0 + off;
    for (int r = 0; r < 8; ++r) {
        int m = mbase + r;
        float v = (ntype[m] == 0) ? (acc0[r] + bias0) : (acc1[r] + bias1);
        snew[(size_t)m * 128 + n] = tanhf(v);
    }
}

// g = relu(state @ Wo1 + bo1)  (K=128, N=512)
__global__ void k_gemm_o1(const unsigned short* sB, const unsigned short* pWo1,
                          const float* bo1, unsigned short* gB) {
    const int KC = 4;                           // 128/32
    int lane = threadIdx.x & 31, w = threadIdx.x >> 5;
    int m0 = blockIdx.x * 16;
    int nt = blockIdx.y * 4 + w;                // 0..31
    int off = (lane < 16) ? 0 : 8;
    const unsigned short* arow = sB + (size_t)(m0 + (lane & 15)) * 128 + off;
    const unsigned short* bp   = pWo1 + (size_t)nt * KC * 512 + lane * 16;
    v8f acc = {};
    for (int kc = 0; kc < KC; ++kc) {
        Frag a, b;
        a.f4[0] = *(const v4f*)(arow + kc * 32);
        a.f4[1] = *(const v4f*)(arow + kc * 32 + 16);
        b.f4[0] = *(const v4f*)(bp + (size_t)kc * 512);
        b.f4[1] = *(const v4f*)(bp + (size_t)kc * 512 + 8);
        acc = WMMA_BF16(a.bf, b.bf, acc);
    }
    int n = nt * 16 + (lane & 15);
    float bias = bo1[n];
    int mbase = m0 + off;
    for (int r = 0; r < 8; ++r) {
        int m = mbase + r;
        float v = acc[r] + bias; v = v > 0.0f ? v : 0.0f;
        gB[(size_t)m * 512 + n] = f2bf(v);
    }
}

// out = g @ Wo2 + bo2  (K=512, N=16), f32 output
__global__ void k_gemm_o2(const unsigned short* gB, const unsigned short* pWo2,
                          const float* bo2, float* out) {
    const int KC = 16;                          // 512/32
    int lane = threadIdx.x & 31;
    int m0 = blockIdx.x * 16;
    int off = (lane < 16) ? 0 : 8;
    const unsigned short* arow = gB + (size_t)(m0 + (lane & 15)) * 512 + off;
    const unsigned short* bp   = pWo2 + lane * 16;
    v8f acc = {};
    for (int kc = 0; kc < KC; ++kc) {
        Frag a, b;
        a.f4[0] = *(const v4f*)(arow + kc * 32);
        a.f4[1] = *(const v4f*)(arow + kc * 32 + 16);
        b.f4[0] = *(const v4f*)(bp + (size_t)kc * 512);
        b.f4[1] = *(const v4f*)(bp + (size_t)kc * 512 + 8);
        acc = WMMA_BF16(a.bf, b.bf, acc);
    }
    int n = lane & 15;
    float bias = bo2[n];
    int mbase = m0 + off;
    for (int r = 0; r < 8; ++r)
        out[(size_t)(mbase + r) * 16 + n] = acc[r] + bias;
}

// ---------------------------------------------------------------------------
// Host launcher
// ---------------------------------------------------------------------------
extern "C" void kernel_launch(void* const* d_in, const int* in_sizes, int n_in,
                              void* d_out, int out_size, void* d_ws, size_t ws_size,
                              hipStream_t stream) {
    const float* nodes  = (const float*)d_in[0];
    const float* arcs   = (const float*)d_in[1];
    const float* state0 = (const float*)d_in[2];
    const int*   snd    = (const int*)d_in[3];
    const int*   rcv    = (const int*)d_in[4];
    const int*   ntype  = (const int*)d_in[5];
    const float* Wo1    = (const float*)d_in[6];
    const float* bo1    = (const float*)d_in[7];
    const float* Wo2    = (const float*)d_in[8];
    const float* bo2    = (const float*)d_in[9];
    const float* W1_0   = (const float*)d_in[10];
    const float* b1_0   = (const float*)d_in[11];
    const float* W2_0   = (const float*)d_in[12];
    const float* b2_0   = (const float*)d_in[13];
    const float* W1_1   = (const float*)d_in[14];
    const float* b1_1   = (const float*)d_in[15];
    const float* W2_1   = (const float*)d_in[16];
    const float* W2b_1  = (const float*)d_in[17];   // b2_1

    const int N = in_sizes[5];     // 40000
    const int E = in_sizes[3];     // 640000
    const int MT = N / 16;         // 2500 m-tiles

    // Workspace partition (256B aligned)
    char* ws = (char*)d_ws;
    size_t off = 0;
    auto alloc = [&](size_t bytes) -> void* {
        void* p = ws + off;
        off = (off + bytes + 255) & ~(size_t)255;
        return p;
    };
    float* s_cur = (float*)alloc((size_t)N * 128 * 4);
    float* s_old = (float*)alloc((size_t)N * 128 * 4);
    float* s_new = (float*)alloc((size_t)N * 128 * 4);
    float* aggs  = (float*)alloc((size_t)N * 128 * 4);
    unsigned short* inpB = (unsigned short*)alloc((size_t)N * 672 * 2);
    unsigned short* h0B  = (unsigned short*)alloc((size_t)N * 512 * 2);
    unsigned short* h1B  = (unsigned short*)alloc((size_t)N * 512 * 2);
    unsigned short* sB   = (unsigned short*)alloc((size_t)N * 128 * 2);
    unsigned short* gB   = (unsigned short*)alloc((size_t)N * 512 * 2);
    unsigned short* pW10 = (unsigned short*)alloc((size_t)672 * 512 * 2);
    unsigned short* pW11 = (unsigned short*)alloc((size_t)672 * 512 * 2);
    unsigned short* pW20 = (unsigned short*)alloc((size_t)512 * 128 * 2);
    unsigned short* pW21 = (unsigned short*)alloc((size_t)512 * 128 * 2);
    unsigned short* pWo1 = (unsigned short*)alloc((size_t)128 * 512 * 2);
    unsigned short* pWo2 = (unsigned short*)alloc((size_t)512 * 16 * 2);
    int* flags = (int*)alloc(4 * sizeof(int));
    // aggc (N*288 f32 = 46MB) aliases h0B+h1B (82MB): only live during setup,
    // before h0B/h1B are first written; rebuilt from scratch on every launch.
    float* aggc = (float*)h0B;

    auto grd = [](long total, int bs) { return (unsigned)((total + bs - 1) / bs); };

    // ---- setup ----
    k_zero_i32<<<1, 32, 0, stream>>>(flags, 4);
    k_zero_f32<<<grd((long)N * 288, 256), 256, 0, stream>>>(aggc, N * 288, nullptr);
    k_init_state<<<grd((long)N * 128, 256), 256, 0, stream>>>(state0, s_cur, s_old, N * 128);

    k_pack_b<<<grd(672L * 512, 256), 256, 0, stream>>>(W1_0, pW10, 672, 512, 672 * 512);
    k_pack_b<<<grd(672L * 512, 256), 256, 0, stream>>>(W1_1, pW11, 672, 512, 672 * 512);
    k_pack_b<<<grd(512L * 128, 256), 256, 0, stream>>>(W2_0, pW20, 512, 128, 512 * 128);
    k_pack_b<<<grd(512L * 128, 256), 256, 0, stream>>>(W2_1, pW21, 512, 128, 512 * 128);
    k_pack_b<<<grd(128L * 512, 256), 256, 0, stream>>>(Wo1,  pWo1, 128, 512, 128 * 512);
    k_pack_b<<<grd(512L * 16,  256), 256, 0, stream>>>(Wo2,  pWo2, 512, 16,  512 * 16);

    k_scatter_comp_nodes<<<grd((long)E * 128, 256), 256, 0, stream>>>(nodes, snd, rcv, ntype, aggc, E * 128);
    k_scatter_comp_arcs<<<grd((long)E * 32, 256), 256, 0, stream>>>(arcs, rcv, aggc, E * 32);
    k_build_static<<<grd((long)N * 416, 256), 256, 0, stream>>>(nodes, aggc, inpB, N * 416);

    // ---- convergence loop (MAX_ITER = 4, per-iteration any() condition) ----
    for (int it = 0; it < 4; ++it) {
        const int* fl = flags + it;
        k_flag<<<grd(N, 256), 256, 0, stream>>>(s_cur, s_old, flags + it, N);
        k_zero_f32<<<grd((long)N * 128, 256), 256, 0, stream>>>(aggs, N * 128, fl);
        k_scatter_state<<<grd((long)E * 128, 256), 256, 0, stream>>>(s_cur, snd, rcv, aggs, E * 128, fl);
        k_build_dyn<<<grd((long)N * 128, 256), 256, 0, stream>>>(s_cur, aggs, inpB, N * 128, fl);
        k_gemm1<<<dim3(MT, 8), 128, 0, stream>>>(inpB, pW10, pW11, b1_0, b1_1, h0B, h1B, fl);
        k_gemm2<<<dim3(MT, 2), 128, 0, stream>>>(h0B, h1B, pW20, pW21, b2_0, W2b_1, ntype, s_new, fl);
        k_commit<<<grd((long)N * 128, 256), 256, 0, stream>>>(s_cur, s_old, s_new, N * 128, fl);
    }

    // ---- output head ----
    k_to_bf16<<<grd((long)N * 128, 256), 256, 0, stream>>>(s_cur, sB, N * 128);
    k_gemm_o1<<<dim3(MT, 8), 128, 0, stream>>>(sB, pWo1, bo1, gB);
    k_gemm_o2<<<dim3(MT, 1), 32, 0, stream>>>(gB, pWo2, bo2, (float*)d_out);
}
